// single_layer_lstm_59287728554363
// MI455X (gfx1250) — compile-verified, matrix operand order checked by reading, not run on hardware
//
#include <hip/hip_runtime.h>
#include <hip/hip_bf16.h>

// ---------------------------------------------------------------------------
// LSTM on gfx1250 (MI455X): bf16 WMMA (v_wmma_f32_16x16x32_bf16), f32 accum.
// B=64, SEQ=512, I=H=1024.
//
// Phase 1 (parallel): P[t] = x_t @ [W_ii|W_if|W_ig|W_io] + bias, fp32,
//   pre-swizzled WMMA-C layout, NT stores. x_t staged once per block in LDS
//   (bf16, WMMA-A order) so each weight tile is read once per timestep.
// Phase 2: ONE persistent kernel (64 WGs x 256 thr), 512 timesteps with a
//   device-scope atomic barrier between steps. c lives in LDS. h ping-pongs
//   through global in pre-swizzled bf16 WMMA-A order. P tiles for t+1 are
//   prefetched before the barrier spin. K-loop is software-pipelined.
// Fallback (small ws): persistent fused variant (x projection each step).
// ---------------------------------------------------------------------------

typedef __attribute__((ext_vector_type(16))) __bf16        v16bf;
typedef __attribute__((ext_vector_type(8)))  float         v8f;
typedef __attribute__((ext_vector_type(4)))  float         v4f;
typedef __attribute__((ext_vector_type(8)))  unsigned int  v8u;
typedef unsigned short u16;

#define BATCH 64
#define SEQ   512
#define ISZ   1024
#define HSZ   1024
#define NWG   64          // persistent grid size

__device__ __forceinline__ unsigned pack2bf(float lo, float hi) {
    unsigned a = __float_as_uint(lo), b = __float_as_uint(hi);
    a = (a + 0x7FFFu + ((a >> 16) & 1u)) >> 16;
    b = (b + 0x7FFFu + ((b >> 16) & 1u)) >> 16;
    return a | (b << 16);
}
__device__ __forceinline__ u16 bf16_val(float f) {
    return (u16)(pack2bf(f, 0.0f) & 0xFFFFu);
}
__device__ __forceinline__ v16bf as_v16bf(v8u u) {
    union { v8u u; v16bf b; } c; c.u = u; return c.b;
}
__device__ __forceinline__ float sigm(float v) {
    return 1.0f / (1.0f + __expf(-v));
}
__device__ __forceinline__ v16bf ldB(const u16* p) {       // 32B tile slice
    return as_v16bf(*(const v8u*)p);
}
// pack two 8-float runs (p[0..7], p[16..23]) into a WMMA-A lane slice
__device__ __forceinline__ v8u packA_f32(const float* p) {
    float4 f0 = *(const float4*)(p);
    float4 f1 = *(const float4*)(p + 4);
    float4 f2 = *(const float4*)(p + 16);
    float4 f3 = *(const float4*)(p + 20);
    v8u au;
    au[0] = pack2bf(f0.x, f0.y); au[1] = pack2bf(f0.z, f0.w);
    au[2] = pack2bf(f1.x, f1.y); au[3] = pack2bf(f1.z, f1.w);
    au[4] = pack2bf(f2.x, f2.y); au[5] = pack2bf(f2.z, f2.w);
    au[6] = pack2bf(f3.x, f3.y); au[7] = pack2bf(f3.z, f3.w);
    return au;
}
__device__ __forceinline__ v8f wmma_bf16(v16bf a, v16bf b, v8f c) {
    return __builtin_amdgcn_wmma_f32_16x16x32_bf16(
               false, a, false, b, (short)0, c, false, false);
}

// ---------------------------------------------------------------------------
// Weights: bf16, K=2048 x N=4096 (n = gate*1024 + h), (nt,kt) 16x32 tiles,
// each tile = 32 lanes x 16 bf16 in WMMA-B per-lane order.
// ---------------------------------------------------------------------------
__global__ __launch_bounds__(256) void pack_w(
    const float* __restrict__ wi0, const float* __restrict__ wh0,
    const float* __restrict__ wi1, const float* __restrict__ wh1,
    const float* __restrict__ wi2, const float* __restrict__ wh2,
    const float* __restrict__ wi3, const float* __restrict__ wh3,
    u16* __restrict__ out)
{
    int tid  = blockIdx.x * 256 + threadIdx.x;     // 0 .. 8388607
    int e    = tid & 15;
    int lane = (tid >> 4) & 31;
    int kt   = (tid >> 9) & 63;
    int nt   = tid >> 15;
    int half = lane >> 4;
    int kk   = e + 8 * half + 8 * (e >> 3);
    int k    = kt * 32 + kk;
    int n    = nt * 16 + (lane & 15);
    int g    = n >> 10;
    int h    = n & 1023;
    const float* Wi = (g == 0) ? wi0 : (g == 1) ? wi1 : (g == 2) ? wi2 : wi3;
    const float* Wh = (g == 0) ? wh0 : (g == 1) ? wh1 : (g == 2) ? wh2 : wh3;
    float v = (k < ISZ) ? Wi[k * HSZ + h] : Wh[(k - ISZ) * HSZ + h];
    out[tid] = bf16_val(v);
}

__global__ __launch_bounds__(256) void zero_u32(unsigned* __restrict__ p, int n) {
    int i = blockIdx.x * 256 + threadIdx.x;
    if (i < n) p[i] = 0u;
}

// ---------------------------------------------------------------------------
// Phase 1: input projections for all t. Grid (8, 512) = (nt-group, t).
// Block covers the FULL batch (4 M-tiles) x 32 N-tiles: x_t is staged once
// per block into LDS (bf16 WMMA-A order), each weight tile loaded once and
// reused by 4 WMMAs. Wave wv handles 4 N-tiles x 4 M-tiles (16 acc tiles).
// P tile layout: ((t*4+mt)*256 + nt)*256 + lane*8 + r   (WMMA-C swizzle).
// ---------------------------------------------------------------------------
__global__ __launch_bounds__(256) void xproj(
    const float* __restrict__ x, const u16* __restrict__ Wpack,
    const float* __restrict__ bias_i, const float* __restrict__ bias_f,
    const float* __restrict__ bias_g, const float* __restrict__ bias_o,
    float* __restrict__ P)
{
    __shared__ u16 xa[4][32][32][16];   // 128 KB: [mt][kt][lane][e], A order

    const int tid  = threadIdx.x;
    const int lane = tid & 31;
    const int wv   = tid >> 5;
    const int col  = lane & 15;
    const int t    = blockIdx.y;
    const int ntbase = blockIdx.x * 32 + wv * 4;

    // cooperative stage of x_t (64 rows x 1024) -> bf16 WMMA-A tiles in LDS
#pragma unroll 4
    for (int u = 0; u < 16; ++u) {
        int unit = u * 256 + tid;            // (mt, kt, dlane)
        int smt   = unit >> 10;
        int skt   = (unit >> 5) & 31;
        int dlane = unit & 31;
        int srow  = smt * 16 + (dlane & 15);
        int shalf = dlane >> 4;
        const float* p = x + ((size_t)srow * SEQ + t) * ISZ + skt * 32 + 8 * shalf;
        *(v8u*)&xa[smt][skt][dlane][0] = packA_f32(p);
    }
    __syncthreads();

    v8f acc[4][4];                            // [mt][j]
    const u16* wb[4];
#pragma unroll
    for (int j = 0; j < 4; ++j) {
        int n = (ntbase + j) * 16 + col;
        int g = n >> 10, id = n & 1023;
        const float* bp = (g == 0) ? bias_i : (g == 1) ? bias_f
                        : (g == 2) ? bias_g : bias_o;
        float bv = bp[id];
#pragma unroll
        for (int mt = 0; mt < 4; ++mt)
#pragma unroll
            for (int r = 0; r < 8; ++r) acc[mt][j][r] = bv;
        wb[j] = Wpack + (size_t)(ntbase + j) * 64 * 512 + lane * 16;
    }

#pragma unroll 2
    for (int kt = 0; kt < 32; ++kt) {
        v16bf a[4];
#pragma unroll
        for (int mt = 0; mt < 4; ++mt)
            a[mt] = as_v16bf(*(const v8u*)&xa[mt][kt][lane][0]);
#pragma unroll
        for (int j = 0; j < 4; ++j) {
            v16bf b = ldB(wb[j] + kt * 512);
#pragma unroll
            for (int mt = 0; mt < 4; ++mt)
                acc[mt][j] = wmma_bf16(a[mt], b, acc[mt][j]);
        }
    }

#pragma unroll
    for (int mt = 0; mt < 4; ++mt)
#pragma unroll
        for (int j = 0; j < 4; ++j) {
            float* dst = P + ((((size_t)t * 4 + mt) * 256) + ntbase + j) * 256
                           + lane * 8;
            union { v8f v; v4f q[2]; } u; u.v = acc[mt][j];
            __builtin_nontemporal_store(u.q[0], (v4f*)dst);
            __builtin_nontemporal_store(u.q[1], (v4f*)(dst + 4));
        }
}

// ---------------------------------------------------------------------------
// Persistent LSTM scan. Grid: 64 WGs (16 hidden cols each), 256 thr (8 waves).
// Wave w: M-tile mt = w&3, gate pair gp = w>>2 ({i,f} | {g,o}).
// FUSED=0: acc init from prefetched P tile, K loop = 32 recurrent tiles.
// FUSED=1: acc init from bias, + 32 x-tiles (on-the-fly cvt).
// ---------------------------------------------------------------------------
template<int FUSED>
__global__ __launch_bounds__(256) void lstm_persist(
    const float* __restrict__ x, const float* __restrict__ P,
    const u16*   __restrict__ Wpack,
    u16* __restrict__ hbufA, u16* __restrict__ hbufB,
    unsigned* __restrict__ ctr,
    const float* __restrict__ bias_i, const float* __restrict__ bias_f,
    const float* __restrict__ bias_g, const float* __restrict__ bias_o,
    float* __restrict__ out_seq, float* __restrict__ out_hT,
    float* __restrict__ out_cT)
{
    __shared__ float pre[4][BATCH][16];   // 16 KB gate preactivations
    __shared__ float cLds[BATCH][16];     // 4 KB private cell state

    const int tid  = threadIdx.x;
    const int lane = tid & 31;
    const int wv   = tid >> 5;
    const int mt   = wv & 3;
    const int gp   = wv >> 2;
    const int ht   = blockIdx.x;
    const int col  = lane & 15;
    const int half = lane >> 4;

    const int nt0 = (gp * 2 + 0) * 64 + ht;
    const int nt1 = (gp * 2 + 1) * 64 + ht;

    // c0 = 0
#pragma unroll
    for (int e = 0; e < 4; ++e) {
        int flat = e * 256 + tid;
        cLds[flat >> 4][flat & 15] = 0.0f;
    }

    const u16* wb0h = Wpack + ((size_t)nt0 * 64 + 32) * 512 + lane * 16;
    const u16* wb1h = Wpack + ((size_t)nt1 * 64 + 32) * 512 + lane * 16;

    const float b0 = (gp == 0 ? bias_i : bias_g)[ht * 16 + col];
    const float b1 = (gp == 0 ? bias_f : bias_o)[ht * 16 + col];

    // prefetch P tiles for t = 0
    v8f pacc0, pacc1;
    if (!FUSED) {
        const float* p0 = P + ((size_t)mt * 256 + nt0) * 256 + lane * 8;
        const float* p1 = P + ((size_t)mt * 256 + nt1) * 256 + lane * 8;
        pacc0 = __builtin_nontemporal_load((const v8f*)p0);
        pacc1 = __builtin_nontemporal_load((const v8f*)p1);
    }
    __syncthreads();

    for (int t = 0; t < SEQ; ++t) {
        const u16* hcur = (t & 1) ? hbufB : hbufA;
        u16*       hnxt = (t & 1) ? hbufA : hbufB;

        v8f acc0, acc1;
        if (FUSED) {
#pragma unroll
            for (int i = 0; i < 8; ++i) { acc0[i] = b0; acc1[i] = b1; }
            const u16* wb0 = Wpack + (size_t)nt0 * 64 * 512 + lane * 16;
            const u16* wb1 = Wpack + (size_t)nt1 * 64 * 512 + lane * 16;
            const float* xrow =
                x + ((size_t)(mt * 16 + col) * SEQ + t) * ISZ + 8 * half;
#pragma unroll 4
            for (int kt = 0; kt < 32; ++kt) {
                v16bf a = as_v16bf(packA_f32(xrow + kt * 32));
                acc0 = wmma_bf16(a, ldB(wb0 + kt * 512), acc0);
                acc1 = wmma_bf16(a, ldB(wb1 + kt * 512), acc1);
            }
        } else {
            acc0 = pacc0;
            acc1 = pacc1;
        }

        // recurrent half: software-pipelined (prefetch kt+1 before WMMA kt)
        {
            const u16* ha = hcur + ((size_t)mt * 32 * 32 + lane) * 16;
            v8u a_n  = *(const v8u*)(ha);
            v8u b0_n = *(const v8u*)(wb0h);
            v8u b1_n = *(const v8u*)(wb1h);
#pragma unroll
            for (int kt = 0; kt < 32; ++kt) {
                v8u a_c = a_n, b0_c = b0_n, b1_c = b1_n;
                if (kt < 31) {
                    a_n  = *(const v8u*)(ha   + (kt + 1) * 512);
                    b0_n = *(const v8u*)(wb0h + (kt + 1) * 512);
                    b1_n = *(const v8u*)(wb1h + (kt + 1) * 512);
                }
                acc0 = wmma_bf16(as_v16bf(a_c), as_v16bf(b0_c), acc0);
                acc1 = wmma_bf16(as_v16bf(a_c), as_v16bf(b1_c), acc1);
            }
        }

        // exchange gate tiles through LDS (C layout: row = r + 8*half)
#pragma unroll
        for (int r = 0; r < 8; ++r) {
            int row = mt * 16 + r + 8 * half;
            pre[gp * 2 + 0][row][col] = acc0[r];
            pre[gp * 2 + 1][row][col] = acc1[r];
        }

        // prefetch P tiles for t+1 (latency hidden behind barrier)
        if (!FUSED && t + 1 < SEQ) {
            size_t base = (((size_t)(t + 1) * 4 + mt) * 256) * 256 + (size_t)lane * 8;
            pacc0 = __builtin_nontemporal_load((const v8f*)(P + base + (size_t)nt0 * 256));
            pacc1 = __builtin_nontemporal_load((const v8f*)(P + base + (size_t)nt1 * 256));
        }
        __syncthreads();

        // cell update: 1024 (b, col) cells, 4 per thread
#pragma unroll
        for (int e = 0; e < 4; ++e) {
            int flat = e * 256 + tid;
            int b    = flat >> 4;
            int c    = flat & 15;
            int hid  = ht * 16 + c;

            float i_t = sigm(pre[0][b][c]);
            float f_t = sigm(pre[1][b][c]);
            float g_t = tanhf(pre[2][b][c]);
            float o_t = sigm(pre[3][b][c]);
            float cn  = f_t * cLds[b][c] + i_t * g_t;
            float hn  = o_t * tanhf(cn);

            cLds[b][c] = cn;
            __builtin_nontemporal_store(
                hn, out_seq + ((size_t)b * SEQ + t) * HSZ + hid);

            int mt2 = b >> 4, lr = b & 15;
            int kt2 = hid >> 5, kk = hid & 31;
            int lane2 = lr + (((kk >> 3) & 1) << 4);
            int e2    = (kk & 7) + ((kk >> 4) << 3);
            hnxt[(((mt2 * 32 + kt2) * 32 + lane2) << 4) + e2] = bf16_val(hn);

            if (t == SEQ - 1) {
                out_hT[b * HSZ + hid] = hn;
                out_cT[b * HSZ + hid] = cn;
            }
        }

        // device-scope barrier between timesteps (monotonic counter,
        // relaxed polls + one acquire fence on exit)
        if (t != SEQ - 1) {
            __threadfence();
            __syncthreads();
            if (tid == 0) {
                __hip_atomic_fetch_add(ctr, 1u, __ATOMIC_RELEASE,
                                       __HIP_MEMORY_SCOPE_AGENT);
                unsigned target = (unsigned)NWG * (unsigned)(t + 1);
                while (__hip_atomic_load(ctr, __ATOMIC_RELAXED,
                                         __HIP_MEMORY_SCOPE_AGENT) < target) {
                    __builtin_amdgcn_s_sleep(1);
                }
                __builtin_amdgcn_fence(__ATOMIC_ACQUIRE, "agent");
            }
            __syncthreads();
        }
    }
}

// ---------------------------------------------------------------------------
// ws layout (fast):  [Wpack 16MB][P 512MB][hbufA 128K][hbufB 128K][ctr 1K]
// ws layout (small): [Wpack 16MB][hbufA 128K][hbufB 128K][ctr 1K]
// ---------------------------------------------------------------------------
extern "C" void kernel_launch(void* const* d_in, const int* in_sizes, int n_in,
                              void* d_out, int out_size, void* d_ws, size_t ws_size,
                              hipStream_t stream) {
    const float* x    = (const float*)d_in[0];
    const float* W_ii = (const float*)d_in[1];
    const float* W_hi = (const float*)d_in[2];
    const float* b_i  = (const float*)d_in[3];
    const float* W_if = (const float*)d_in[4];
    const float* W_hf = (const float*)d_in[5];
    const float* b_f  = (const float*)d_in[6];
    const float* W_ig = (const float*)d_in[7];
    const float* W_hg = (const float*)d_in[8];
    const float* b_g  = (const float*)d_in[9];
    const float* W_io = (const float*)d_in[10];
    const float* W_ho = (const float*)d_in[11];
    const float* b_o  = (const float*)d_in[12];

    const size_t WPACK_B = (size_t)16 * 1024 * 1024;
    const size_t P_B     = (size_t)512 * 1024 * 1024;
    const size_t STATE_B = (size_t)257 * 1024;
    const bool   fast    = ws_size >= WPACK_B + P_B + STATE_B;

    char*     ws     = (char*)d_ws;
    u16*      Wpack  = (u16*)ws;
    float*    P      = (float*)(ws + WPACK_B);
    char*     st     = ws + WPACK_B + (fast ? P_B : 0);
    u16*      hbufA  = (u16*)st;
    u16*      hbufB  = (u16*)(st + 128 * 1024);
    unsigned* ctr    = (unsigned*)(st + 256 * 1024);

    float* out     = (float*)d_out;
    float* out_seq = out;
    float* out_hT  = out + (size_t)BATCH * SEQ * HSZ;
    float* out_cT  = out_hT + (size_t)BATCH * HSZ;

    pack_w<<<32768, 256, 0, stream>>>(W_ii, W_hi, W_if, W_hf,
                                      W_ig, W_hg, W_io, W_ho, Wpack);
    // zero hbufA + hbufB + ctr (ctr MUST be re-zeroed every call: graph replay)
    zero_u32<<<258, 256, 0, stream>>>((unsigned*)st, (256 * 1024 + 1024) / 4);

    if (fast) {
        xproj<<<dim3(8, SEQ), 256, 0, stream>>>(x, Wpack, b_i, b_f, b_g, b_o, P);
        lstm_persist<0><<<NWG, 256, 0, stream>>>(x, P, Wpack, hbufA, hbufB, ctr,
                                                 b_i, b_f, b_g, b_o,
                                                 out_seq, out_hT, out_cT);
    } else {
        lstm_persist<1><<<NWG, 256, 0, stream>>>(x, P, Wpack, hbufA, hbufB, ctr,
                                                 b_i, b_f, b_g, b_o,
                                                 out_seq, out_hT, out_cT);
    }
}
